// MultiBodySE3HamNODE_11828339933650
// MI455X (gfx1250) — compile-verified, matrix operand order checked by reading, not run on hardware
//
#include <hip/hip_runtime.h>
#include <hip/hip_bf16.h>
#include <cmath>

// ---------------------------------------------------------------------------
// MultiBodySE3HamNODE forward pass on MI455X (gfx1250).
// Heavy math = bf16 WMMA GEMMs (v_wmma_f32_16x16x32_bf16, f32 accumulate).
// Tile staging uses CDNA5 async global->LDS copies (ASYNCcnt tracked).
// Weights are stored n-major (pre-transposed) so all LDS staging is b128.
// ---------------------------------------------------------------------------

typedef __bf16 bf16_t;
typedef __attribute__((ext_vector_type(16))) __bf16 v16bf;
typedef __attribute__((ext_vector_type(8)))  __bf16 v8bf;
typedef __attribute__((ext_vector_type(8)))  float  v8f;

#define BATCH  8192
#define NBODY  4
#define BN     (BATCH * NBODY)      // 32768
#define ROW    78                   // N*18 + UDIM
#define HID    400
#define KH     416                  // pad32(HID)  (K stride)
#define NH     448                  // pad64(HID)  (row count of n-major weights)
#define UD     6
#define EPSI   0.01f

enum { EPI_LIN = 0, EPI_TANH = 1, EPI_TANHBWD = 2 };

// ---------------------------------------------------------------------------
// CDNA5 async global->LDS 16-byte copy (per lane), tracked by ASYNCcnt.
// LDS operand = wave-relative LDS byte offset = low 32 bits of generic addr
// (shared aperture occupies only the high 32 bits on gfx1250).
// ---------------------------------------------------------------------------
__device__ __forceinline__ void async_b128(void* lds, const void* gptr) {
  unsigned l = (unsigned)(size_t)lds;
  asm volatile("global_load_async_to_lds_b128 %0, %1, off"
               :: "v"(l), "v"(gptr) : "memory");
}
__device__ __forceinline__ void wait_async0() {
  asm volatile("s_wait_asynccnt 0x0" ::: "memory");
}

// ---------------------------------------------------------------------------
// WMMA fragment load from LDS.  Tile stored [major][32] (bf16, k contiguous).
// CDNA5 16-bit A layout: lanes 0-15 -> rows, K 0..7 (v0..v3) + 16..23 (v4..v7);
// lanes 16-31 -> same rows, K 8..15 / 24..31.  B mirrors with N major.
// ---------------------------------------------------------------------------
__device__ __forceinline__ v16bf ldsfrag(const bf16_t* base, int major0, int lane) {
  const bf16_t* p = base + (size_t)(major0 + (lane & 15)) * 32 + ((lane >> 4) << 3);
  v8bf lo = *(const v8bf*)(p);
  v8bf hi = *(const v8bf*)(p + 16);
  v16bf f;
#pragma unroll
  for (int j = 0; j < 8; ++j) { f[j] = lo[j]; f[j + 8] = hi[j]; }
  return f;
}

template<int EPI, bool HAS_BIAS>
__device__ __forceinline__ void epi_store(
    v8f acc, int row0, int col,
    const float* __restrict__ bias, const bf16_t* __restrict__ H, int ldh,
    bf16_t* __restrict__ Cb, int ldcb, float* __restrict__ Cf, int ldcf, int Nout)
{
  if (col >= Nout) return;
#pragma unroll
  for (int r = 0; r < 8; ++r) {
    int row = row0 + r;
    float v = acc[r];
    if (HAS_BIAS) v += bias[col];
    if (EPI == EPI_TANH) v = tanhf(v);
    if (EPI == EPI_TANHBWD) {
      float h = (float)H[(size_t)row * ldh + col];
      v *= (1.0f - h * h);
    }
    if (Cb) Cb[(size_t)row * ldcb + col] = (bf16_t)v;
    if (Cf) Cf[(size_t)row * ldcf + col] = v;
  }
}

// ---------------------------------------------------------------------------
// Fused GEMM:  C[M x Nout] = epi(A[M x Kp] @ Wt^T + bias)
// A  : bf16 row-major, stride lda = Kp (multiple of 32)
// Wt : bf16 n-major (i.e. W transposed), rows pad64(Nout), stride ldw = Kp
// M multiple of 128.  Block tile 128x64, 8 waves, 32x32 per wave.
// ---------------------------------------------------------------------------
template<int EPI, bool HAS_BIAS>
__global__ __launch_bounds__(256) void gemm_wmma(
    const bf16_t* __restrict__ A, int lda,
    const bf16_t* __restrict__ W, int ldw,
    const float* __restrict__ bias,
    const bf16_t* __restrict__ H, int ldh,
    bf16_t* __restrict__ Cb, int ldcb,
    float* __restrict__ Cf, int ldcf,
    int M, int Kp, int Nout)
{
  __shared__ __align__(16) bf16_t sA[128 * 32];
  __shared__ __align__(16) bf16_t sB[64 * 32];   // [n][k]

  const int tid  = threadIdx.x;
  const int lane = tid & 31;
  const int wid  = tid >> 5;
  const int wm   = (wid & 3) * 32;   // wave M offset in tile
  const int wn   = (wid >> 2) * 32;  // wave N offset in tile
  const int m0   = blockIdx.y * 128;
  const int n0   = blockIdx.x * 64;

  // staging coordinates (constant per thread)
  const int ar0 = tid >> 2;                 // A rows: tid/4 and +64
  const int akc = (tid & 3) << 3;           // A k-chunk (8 bf16 = 16B)
  const int wn0 = tid >> 2;                 // W n-row
  const int wkc = (tid & 3) << 3;           // W k-chunk

  v8f c00 = {0.f,0.f,0.f,0.f,0.f,0.f,0.f,0.f};
  v8f c01 = c00, c10 = c00, c11 = c00;

  for (int k0 = 0; k0 < Kp; k0 += 32) {
    // async-stage A tile (128x32) : 2 x b128 per thread
    async_b128(&sA[ar0 * 32 + akc],
               A + (size_t)(m0 + ar0) * lda + k0 + akc);
    async_b128(&sA[(ar0 + 64) * 32 + akc],
               A + (size_t)(m0 + ar0 + 64) * lda + k0 + akc);
    // async-stage W tile (64 n-rows x 32 k) : 1 x b128 per thread
    async_b128(&sB[wn0 * 32 + wkc],
               W + (size_t)(n0 + wn0) * ldw + k0 + wkc);
    if (k0 + 32 < Kp)   // global_prefetch_b8 of next A tile
      __builtin_prefetch(A + (size_t)(m0 + ar0) * lda + k0 + 32 + akc, 0, 1);

    wait_async0();        // this wave's async copies landed in LDS
    __syncthreads();      // everyone's copies visible

    v16bf a0 = ldsfrag(sA, wm,      lane);
    v16bf a1 = ldsfrag(sA, wm + 16, lane);
    v16bf b0 = ldsfrag(sB, wn,      lane);
    v16bf b1 = ldsfrag(sB, wn + 16, lane);

    c00 = __builtin_amdgcn_wmma_f32_16x16x32_bf16(false, a0, false, b0, (short)0, c00, false, false);
    c01 = __builtin_amdgcn_wmma_f32_16x16x32_bf16(false, a0, false, b1, (short)0, c01, false, false);
    c10 = __builtin_amdgcn_wmma_f32_16x16x32_bf16(false, a1, false, b0, (short)0, c10, false, false);
    c11 = __builtin_amdgcn_wmma_f32_16x16x32_bf16(false, a1, false, b1, (short)0, c11, false, false);
    __syncthreads();
  }

  const int rsub = (lane >> 4) << 3;   // C layout: lanes>=16 hold M rows +8
  const int csub = lane & 15;
  epi_store<EPI,HAS_BIAS>(c00, m0+wm+rsub,    n0+wn+csub,    bias,H,ldh,Cb,ldcb,Cf,ldcf,Nout);
  epi_store<EPI,HAS_BIAS>(c01, m0+wm+rsub,    n0+wn+16+csub, bias,H,ldh,Cb,ldcb,Cf,ldcf,Nout);
  epi_store<EPI,HAS_BIAS>(c10, m0+wm+16+rsub, n0+wn+csub,    bias,H,ldh,Cb,ldcb,Cf,ldcf,Nout);
  epi_store<EPI,HAS_BIAS>(c11, m0+wm+16+rsub, n0+wn+16+csub, bias,H,ldh,Cb,ldcb,Cf,ldcf,Nout);
}

// ---------------------------------------------------------------------------
// Elementwise / packing kernels
// ---------------------------------------------------------------------------
__device__ __forceinline__ void cross3(const float* a, const float* b, float* o) {
  o[0] = a[1]*b[2] - a[2]*b[1];
  o[1] = a[2]*b[0] - a[0]*b[2];
  o[2] = a[0]*b[1] - a[1]*b[0];
}

// n-major (transposed) weight conversion: out[n*ldk + k] = W[k*N + n]
__global__ void k_w_nmajor(const float* __restrict__ W, bf16_t* __restrict__ out,
                           int K, int N, int NN, int ldk) {
  int idx = blockIdx.x * blockDim.x + threadIdx.x;
  if (idx >= NN * ldk) return;
  int n = idx / ldk, k = idx % ldk;
  out[idx] = (n < N && k < K) ? (bf16_t)W[k * N + n] : (bf16_t)0.f;
}

// padded row-major copy (this IS the n-major form of W^T): out[r*ldo+c]=W[r*N+c]
__global__ void k_w_pad(const float* __restrict__ W, bf16_t* __restrict__ out,
                        int K, int N, int Rp, int ldo) {
  int idx = blockIdx.x * blockDim.x + threadIdx.x;
  if (idx >= Rp * ldo) return;
  int r = idx / ldo, c = idx % ldo;
  out[idx] = (r < K && c < N) ? (bf16_t)W[r * N + c] : (bf16_t)0.f;
}

__global__ void k_convert_bias(const float* __restrict__ b, float* __restrict__ out,
                               int N, int ldo) {
  int idx = blockIdx.x * blockDim.x + threadIdx.x;
  if (idx >= ldo) return;
  out[idx] = (idx < N) ? b[idx] : 0.f;
}

__global__ void k_pack_xR(const float* __restrict__ inp, bf16_t* __restrict__ Ax,
                          bf16_t* __restrict__ AR) {
  int idx = blockIdx.x * blockDim.x + threadIdx.x;
  if (idx >= BN * 32) return;
  int m = idx >> 5, j = idx & 31;
  int b = m >> 2, n = m & 3;
  const float* base = inp + (size_t)b * ROW + n * 18;
  Ax[idx] = (j < 3) ? (bf16_t)base[j]     : (bf16_t)0.f;
  AR[idx] = (j < 9) ? (bf16_t)base[3 + j] : (bf16_t)0.f;
}

__global__ void k_pack_pose(const float* __restrict__ inp, bf16_t* __restrict__ Ap) {
  int idx = blockIdx.x * blockDim.x + threadIdx.x;
  if (idx >= BATCH * 64) return;
  int b = idx >> 6, c = idx & 63;
  float v = 0.f;
  if (c < 48) { int n = c / 12, j = c % 12; v = inp[(size_t)b * ROW + n * 18 + j]; }
  Ap[idx] = (bf16_t)v;
}

// M = L L^T + eps I ; p = M^{-1} v   (ref naming: "Minv" is this PSD matrix)
__global__ void k_psd_solve(const float* __restrict__ O, const float* __restrict__ inp,
                            int voff, float* __restrict__ Ms, float* __restrict__ p) {
  int m = blockIdx.x * blockDim.x + threadIdx.x;
  if (m >= BN) return;
  const float* o = O + (size_t)m * 64;
  float l00=o[0], l11=o[1], l22=o[2], l10=o[3], l20=o[4], l21=o[5];
  float a00 = l00*l00 + EPSI;
  float a10 = l10*l00;
  float a11 = l10*l10 + l11*l11 + EPSI;
  float a20 = l20*l00;
  float a21 = l20*l10 + l21*l11;
  float a22 = l20*l20 + l21*l21 + l22*l22 + EPSI;
  int b = m >> 2, n = m & 3;
  const float* v = inp + (size_t)b * ROW + n * 18 + voff;
  float c00 = a11*a22 - a21*a21;
  float c01 = a21*a20 - a10*a22;
  float c02 = a10*a21 - a11*a20;
  float id  = 1.f / (a00*c00 + a10*c01 + a20*c02);
  float i00 = c00*id, i01 = c01*id, i02 = c02*id;
  float i11 = (a00*a22 - a20*a20)*id;
  float i12 = (a10*a20 - a00*a21)*id;
  float i22 = (a00*a11 - a10*a10)*id;
  p[m*3+0] = i00*v[0] + i01*v[1] + i02*v[2];
  p[m*3+1] = i01*v[0] + i11*v[1] + i12*v[2];
  p[m*3+2] = i02*v[0] + i12*v[1] + i22*v[2];
  float* s = Ms + (size_t)m * 6;
  s[0]=a00; s[1]=a10; s[2]=a11; s[3]=a20; s[4]=a21; s[5]=a22;
}

// seed for 0.5 p^T (LL^T) p  backprop through head:  dL = (p p^T) L
__global__ void k_quad_seed(const float* __restrict__ O, const float* __restrict__ p,
                            bf16_t* __restrict__ seed) {
  int m = blockIdx.x * blockDim.x + threadIdx.x;
  if (m >= BN) return;
  const float* o = O + (size_t)m * 64;
  float l00=o[0], l11=o[1], l22=o[2], l10=o[3], l20=o[4], l21=o[5];
  float p0=p[m*3+0], p1=p[m*3+1], p2=p[m*3+2];
  float s0 = p0*l00 + p1*l10 + p2*l20;
  float s1 = p1*l11 + p2*l21;
  float s2 = p2*l22;
  float d[6] = { p0*s0, p1*s1, p2*s2, p1*s0, p2*s0, p2*s1 };
  bf16_t z = (bf16_t)0.f;
#pragma unroll
  for (int j = 0; j < 32; ++j) seed[(size_t)m*32 + j] = (j < 6) ? (bf16_t)d[j] : z;
}

// V-net backward seed: d_a2 = (1 - h2^2) * w3[:,0]
__global__ void k_vnet_seed(const bf16_t* __restrict__ h2, const float* __restrict__ w3,
                            bf16_t* __restrict__ seed) {
  int idx = blockIdx.x * blockDim.x + threadIdx.x;
  if (idx >= BATCH * KH) return;
  int c = idx % KH;
  if (c < HID) {
    float h = (float)h2[idx];
    seed[idx] = (bf16_t)((1.f - h*h) * w3[c]);
  } else seed[idx] = (bf16_t)0.f;
}

__global__ void k_force(const float* __restrict__ gq, const float* __restrict__ inp,
                        float* __restrict__ F) {
  int idx = blockIdx.x * blockDim.x + threadIdx.x;
  if (idx >= BATCH * 24) return;
  int b = idx / 24, k = idx % 24;
  const float* u = inp + (size_t)b * ROW + 72;
  const float* g = gq + (size_t)b * 192 + k * 6;
  float acc = 0.f;
#pragma unroll
  for (int j = 0; j < UD; ++j) acc += g[j] * u[j];
  F[idx] = acc;
}

// dx = R vs ; dR_row_i = R_row_i x ws ; also pack bf16 tangents for the JVPs
__global__ void k_dyn1(const float* __restrict__ inp, float* __restrict__ dx,
                       float* __restrict__ dR, bf16_t* __restrict__ dxA,
                       bf16_t* __restrict__ dRA) {
  int m = blockIdx.x * blockDim.x + threadIdx.x;
  if (m >= BN) return;
  int b = m >> 2, n = m & 3;
  const float* base = inp + (size_t)b * ROW + n * 18;
  float R[9], vs[3], ws[3];
#pragma unroll
  for (int i = 0; i < 9; ++i) R[i] = base[3 + i];
#pragma unroll
  for (int i = 0; i < 3; ++i) { vs[i] = base[12 + i]; ws[i] = base[15 + i]; }
  float dxv[3], dRv[9];
#pragma unroll
  for (int i = 0; i < 3; ++i)
    dxv[i] = R[i*3+0]*vs[0] + R[i*3+1]*vs[1] + R[i*3+2]*vs[2];
#pragma unroll
  for (int r = 0; r < 3; ++r) cross3(&R[r*3], ws, &dRv[r*3]);
#pragma unroll
  for (int i = 0; i < 3; ++i) dx[m*3+i] = dxv[i];
#pragma unroll
  for (int i = 0; i < 9; ++i) dR[m*9+i] = dRv[i];
  bf16_t z = (bf16_t)0.f;
#pragma unroll
  for (int j = 0; j < 32; ++j) {
    dxA[(size_t)m*32 + j] = (j < 3) ? (bf16_t)dxv[j] : z;
    dRA[(size_t)m*32 + j] = (j < 9) ? (bf16_t)dRv[j] : z;
  }
}

__global__ void k_combine(const float* __restrict__ dVdq, const float* __restrict__ q1,
                          const float* __restrict__ q2, float* __restrict__ dHdx,
                          float* __restrict__ dHdR) {
  int m = blockIdx.x * blockDim.x + threadIdx.x;
  if (m >= BN) return;
  int b = m >> 2, n = m & 3;
  const float* dv = dVdq + (size_t)b * 64 + n * 12;
#pragma unroll
  for (int i = 0; i < 3; ++i) dHdx[m*3+i] = dv[i] + q1[(size_t)m*32 + i];
#pragma unroll
  for (int i = 0; i < 9; ++i) dHdR[m*9+i] = dv[3+i] + q2[(size_t)m*64 + i];
}

__global__ void k_dpvw(const float* __restrict__ inp, const float* __restrict__ pv,
                       const float* __restrict__ pw, const float* __restrict__ dHdx,
                       const float* __restrict__ dHdR, const float* __restrict__ F,
                       float* __restrict__ dpv, float* __restrict__ dpw) {
  int m = blockIdx.x * blockDim.x + threadIdx.x;
  if (m >= BN) return;
  int b = m >> 2, n = m & 3;
  const float* base = inp + (size_t)b * ROW + n * 18;
  float R[9], vs[3], ws[3], Pv[3], Pw[3], hx[3], hR[9];
#pragma unroll
  for (int i = 0; i < 9; ++i) { R[i] = base[3+i]; hR[i] = dHdR[m*9+i]; }
#pragma unroll
  for (int i = 0; i < 3; ++i) {
    vs[i] = base[12+i]; ws[i] = base[15+i];
    Pv[i] = pv[m*3+i];  Pw[i] = pw[m*3+i]; hx[i] = dHdx[m*3+i];
  }
  const float* Fb = F + (size_t)b * 24 + n * 6;
  float c1[3], c2[3], c3[3], t[3], rc[3] = {0.f,0.f,0.f};
  cross3(Pv, ws, c1);
  cross3(Pw, ws, c2);
  cross3(Pv, vs, c3);
#pragma unroll
  for (int r = 0; r < 3; ++r) {
    cross3(&R[r*3], &hR[r*3], t);
    rc[0] += t[0]; rc[1] += t[1]; rc[2] += t[2];
  }
#pragma unroll
  for (int i = 0; i < 3; ++i) {
    float rtx = R[0*3+i]*hx[0] + R[1*3+i]*hx[1] + R[2*3+i]*hx[2];  // R^T dHdx
    dpv[m*3+i] = c1[i] - rtx + Fb[i];
    dpw[m*3+i] = c2[i] + c3[i] + rc[i] + Fb[3+i];
  }
}

// dv = M dpv + (dL L^T + L dL^T) pv
__global__ void k_jvp_fin(const float* __restrict__ to, const float* __restrict__ O,
                          const float* __restrict__ Ms, const float* __restrict__ dp,
                          const float* __restrict__ p, float* __restrict__ outv) {
  int m = blockIdx.x * blockDim.x + threadIdx.x;
  if (m >= BN) return;
  const float* o = O + (size_t)m * 64;
  const float* t = to + (size_t)m * 64;
  float L[3][3] = {{o[0],0.f,0.f},{o[3],o[1],0.f},{o[4],o[5],o[2]}};
  float D[3][3] = {{t[0],0.f,0.f},{t[3],t[1],0.f},{t[4],t[5],t[2]}};
  const float* s = Ms + (size_t)m * 6;
  float Mm[3][3] = {{s[0],s[1],s[3]},{s[1],s[2],s[4]},{s[3],s[4],s[5]}};
  float Pv[3] = {p[m*3+0], p[m*3+1], p[m*3+2]};
  float Dq[3] = {dp[m*3+0], dp[m*3+1], dp[m*3+2]};
#pragma unroll
  for (int i = 0; i < 3; ++i) {
    float acc = 0.f;
#pragma unroll
    for (int j = 0; j < 3; ++j) {
      float dM = 0.f;
#pragma unroll
      for (int k = 0; k < 3; ++k) dM += D[i][k]*L[j][k] + L[i][k]*D[j][k];
      acc += Mm[i][j] * Dq[j] + dM * Pv[j];
    }
    outv[m*3+i] = acc;
  }
}

__global__ void k_assemble(const float* __restrict__ dx, const float* __restrict__ dR,
                           const float* __restrict__ dv, const float* __restrict__ dw,
                           float* __restrict__ out) {
  int idx = blockIdx.x * blockDim.x + threadIdx.x;
  if (idx >= BATCH * ROW) return;
  int b = idx / ROW, c = idx % ROW;
  float v = 0.f;
  if (c < 72) {
    int n = c / 18, j = c % 18, m = b * 4 + n;
    if      (j < 3)  v = dx[m*3 + j];
    else if (j < 12) v = dR[m*9 + (j-3)];
    else if (j < 15) v = dv[m*3 + (j-12)];
    else             v = dw[m*3 + (j-15)];
  }
  out[idx] = v;
}

// ---------------------------------------------------------------------------
// Host orchestration
// ---------------------------------------------------------------------------
static void gemm_call(hipStream_t s, int epi, bool hasb,
                      const bf16_t* A, int lda, const bf16_t* W, int ldw,
                      const float* bias, const bf16_t* H, int ldh,
                      bf16_t* Cb, int ldcb, float* Cf, int ldcf,
                      int M, int Kp, int Nout) {
  dim3 g((unsigned)((Nout + 63) / 64), (unsigned)(M / 128)), blk(256);
  if (epi == EPI_TANH)
    gemm_wmma<EPI_TANH, true><<<g, blk, 0, s>>>(A,lda,W,ldw,bias,H,ldh,Cb,ldcb,Cf,ldcf,M,Kp,Nout);
  else if (epi == EPI_TANHBWD)
    gemm_wmma<EPI_TANHBWD, false><<<g, blk, 0, s>>>(A,lda,W,ldw,bias,H,ldh,Cb,ldcb,Cf,ldcf,M,Kp,Nout);
  else if (hasb)
    gemm_wmma<EPI_LIN, true><<<g, blk, 0, s>>>(A,lda,W,ldw,bias,H,ldh,Cb,ldcb,Cf,ldcf,M,Kp,Nout);
  else
    gemm_wmma<EPI_LIN, false><<<g, blk, 0, s>>>(A,lda,W,ldw,bias,H,ldh,Cb,ldcb,Cf,ldcf,M,Kp,Nout);
}

#define EL(kern, n, ...) kern<<<((int)(n) + 255) / 256, 256, 0, stream>>>(__VA_ARGS__)

extern "C" void kernel_launch(void* const* d_in, const int* in_sizes, int n_in,
                              void* d_out, int out_size, void* d_ws, size_t ws_size,
                              hipStream_t stream) {
  (void)in_sizes; (void)n_in; (void)out_size; (void)ws_size;
  const float* inp = (const float*)d_in[1];
  const float* m1w[4] = {(const float*)d_in[2], (const float*)d_in[4], (const float*)d_in[6], (const float*)d_in[8]};
  const float* m1b[4] = {(const float*)d_in[3], (const float*)d_in[5], (const float*)d_in[7], (const float*)d_in[9]};
  const float* m2w[4] = {(const float*)d_in[10], (const float*)d_in[12], (const float*)d_in[14], (const float*)d_in[16]};
  const float* m2b[4] = {(const float*)d_in[11], (const float*)d_in[13], (const float*)d_in[15], (const float*)d_in[17]};
  const float* vw[3]  = {(const float*)d_in[18], (const float*)d_in[20], (const float*)d_in[22]};
  const float* vb[2]  = {(const float*)d_in[19], (const float*)d_in[21]};
  const float* gw[3]  = {(const float*)d_in[24], (const float*)d_in[26], (const float*)d_in[28]};
  const float* gb[3]  = {(const float*)d_in[25], (const float*)d_in[27], (const float*)d_in[29]};
  float* out = (float*)d_out;

  char* wsb = (char*)d_ws;
  size_t cur = 0;
  auto B16 = [&](size_t e) { bf16_t* p = (bf16_t*)(wsb + cur); cur += (e*2 + 255) & ~(size_t)255; return p; };
  auto F32 = [&](size_t e) { float*  p = (float*)(wsb + cur);  cur += (e*4 + 255) & ~(size_t)255; return p; };

  // --- forward weights, n-major: [pad64(Nout)] rows x [pad32(Kin)] stride ---
  bf16_t* wM1[4] = { B16(NH*32), B16(NH*KH), B16(NH*KH), B16(64*KH) };
  bf16_t* wM2[4] = { B16(NH*32), B16(NH*KH), B16(NH*KH), B16(64*KH) };
  bf16_t* wV[2]  = { B16(NH*64), B16(NH*KH) };
  bf16_t* wG[3]  = { B16(NH*64), B16(NH*KH), B16(192*KH) };
  // --- backward weights (W^T in n-major == plain padded W) ---
  bf16_t* tM1[4] = { B16(64*KH), B16(NH*KH), B16(NH*KH), B16(NH*32) };
  bf16_t* tM2[4] = { B16(64*KH), B16(NH*KH), B16(NH*KH), B16(NH*32) };
  bf16_t* tV[2]  = { B16(64*KH), B16(NH*KH) };
  // --- padded biases ---
  float* pbM1[4] = { F32(NH), F32(NH), F32(NH), F32(64) };
  float* pbM2[4] = { F32(NH), F32(NH), F32(NH), F32(64) };
  float* pbV[2]  = { F32(NH), F32(NH) };
  float* pbG[3]  = { F32(NH), F32(NH), F32(192) };
  // --- activations & temporaries ---
  bf16_t* A0x = B16((size_t)BN*32);  bf16_t* A0R = B16((size_t)BN*32);
  bf16_t* Apos = B16((size_t)BATCH*64);
  bf16_t* m1h[3] = { B16((size_t)BN*KH), B16((size_t)BN*KH), B16((size_t)BN*KH) };
  bf16_t* m2h[3] = { B16((size_t)BN*KH), B16((size_t)BN*KH), B16((size_t)BN*KH) };
  bf16_t* bt1 = B16((size_t)BN*KH);  bf16_t* bt2 = B16((size_t)BN*KH);
  bf16_t* vh1 = B16((size_t)BATCH*KH); bf16_t* vh2 = B16((size_t)BATCH*KH);
  bf16_t* vseed = B16((size_t)BATCH*KH); bf16_t* vtmp = B16((size_t)BATCH*KH);
  bf16_t* gh1 = B16((size_t)BATCH*KH); bf16_t* gh2 = B16((size_t)BATCH*KH);
  bf16_t* qseed = B16((size_t)BN*32);
  bf16_t* dxA = B16((size_t)BN*32);  bf16_t* dRA = B16((size_t)BN*32);
  float* o1f = F32((size_t)BN*64);   float* o2f = F32((size_t)BN*64);
  float* gqf = F32((size_t)BATCH*192);
  float* Ms1 = F32((size_t)BN*6);    float* Ms2 = F32((size_t)BN*6);
  float* pv  = F32((size_t)BN*3);    float* pw  = F32((size_t)BN*3);
  float* Ff  = F32((size_t)BATCH*24);
  float* dVdq = F32((size_t)BATCH*64);
  float* q1f = F32((size_t)BN*32);   float* q2f = F32((size_t)BN*64);
  float* dxf = F32((size_t)BN*3);    float* dRf = F32((size_t)BN*9);
  float* dHdx = F32((size_t)BN*3);   float* dHdR = F32((size_t)BN*9);
  float* tof = F32((size_t)BN*64);
  float* dpv = F32((size_t)BN*3);    float* dpw = F32((size_t)BN*3);
  float* dvf = F32((size_t)BN*3);    float* dwf = F32((size_t)BN*3);

  // ---- forward weight conversion (n-major, zero padded) ----
  EL(k_w_nmajor, NH*32,  m1w[0], wM1[0], 3,   HID, NH,  32);
  EL(k_w_nmajor, NH*KH,  m1w[1], wM1[1], HID, HID, NH,  KH);
  EL(k_w_nmajor, NH*KH,  m1w[2], wM1[2], HID, HID, NH,  KH);
  EL(k_w_nmajor, 64*KH,  m1w[3], wM1[3], HID, 6,   64,  KH);
  EL(k_w_nmajor, NH*32,  m2w[0], wM2[0], 9,   HID, NH,  32);
  EL(k_w_nmajor, NH*KH,  m2w[1], wM2[1], HID, HID, NH,  KH);
  EL(k_w_nmajor, NH*KH,  m2w[2], wM2[2], HID, HID, NH,  KH);
  EL(k_w_nmajor, 64*KH,  m2w[3], wM2[3], HID, 6,   64,  KH);
  EL(k_w_nmajor, NH*64,  vw[0],  wV[0],  48,  HID, NH,  64);
  EL(k_w_nmajor, NH*KH,  vw[1],  wV[1],  HID, HID, NH,  KH);
  EL(k_w_nmajor, NH*64,  gw[0],  wG[0],  48,  HID, NH,  64);
  EL(k_w_nmajor, NH*KH,  gw[1],  wG[1],  HID, HID, NH,  KH);
  EL(k_w_nmajor, 192*KH, gw[2],  wG[2],  HID, 144, 192, KH);
  // ---- backward weight conversion (padded copies) ----
  EL(k_w_pad, 64*KH, m1w[0], tM1[0], 3,   HID, 64, KH);
  EL(k_w_pad, NH*KH, m1w[1], tM1[1], HID, HID, NH, KH);
  EL(k_w_pad, NH*KH, m1w[2], tM1[2], HID, HID, NH, KH);
  EL(k_w_pad, NH*32, m1w[3], tM1[3], HID, 6,   NH, 32);
  EL(k_w_pad, 64*KH, m2w[0], tM2[0], 9,   HID, 64, KH);
  EL(k_w_pad, NH*KH, m2w[1], tM2[1], HID, HID, NH, KH);
  EL(k_w_pad, NH*KH, m2w[2], tM2[2], HID, HID, NH, KH);
  EL(k_w_pad, NH*32, m2w[3], tM2[3], HID, 6,   NH, 32);
  EL(k_w_pad, 64*KH, vw[0],  tV[0],  48,  HID, 64, KH);
  EL(k_w_pad, NH*KH, vw[1],  tV[1],  HID, HID, NH, KH);
  // ---- biases ----
  for (int i = 0; i < 3; ++i) { EL(k_convert_bias, NH, m1b[i], pbM1[i], HID, NH);
                                EL(k_convert_bias, NH, m2b[i], pbM2[i], HID, NH); }
  EL(k_convert_bias, 64, m1b[3], pbM1[3], 6, 64);
  EL(k_convert_bias, 64, m2b[3], pbM2[3], 6, 64);
  EL(k_convert_bias, NH, vb[0], pbV[0], HID, NH);
  EL(k_convert_bias, NH, vb[1], pbV[1], HID, NH);
  EL(k_convert_bias, NH, gb[0], pbG[0], HID, NH);
  EL(k_convert_bias, NH, gb[1], pbG[1], HID, NH);
  EL(k_convert_bias, 192, gb[2], pbG[2], 144, 192);

  // ---- input packing ----
  EL(k_pack_xR,   BN*32,    inp, A0x, A0R);
  EL(k_pack_pose, BATCH*64, inp, Apos);

  // ---- PSD nets forward (M1: xs, M2: Rs) ----
  gemm_call(stream, EPI_TANH, true, A0x, 32, wM1[0], 32, pbM1[0], nullptr,0, m1h[0],KH, nullptr,0, BN, 32, HID);
  gemm_call(stream, EPI_TANH, true, m1h[0],KH, wM1[1], KH, pbM1[1], nullptr,0, m1h[1],KH, nullptr,0, BN, KH, HID);
  gemm_call(stream, EPI_TANH, true, m1h[1],KH, wM1[2], KH, pbM1[2], nullptr,0, m1h[2],KH, nullptr,0, BN, KH, HID);
  gemm_call(stream, EPI_LIN,  true, m1h[2],KH, wM1[3], KH, pbM1[3], nullptr,0, nullptr,0, o1f,64, BN, KH, 6);
  gemm_call(stream, EPI_TANH, true, A0R, 32, wM2[0], 32, pbM2[0], nullptr,0, m2h[0],KH, nullptr,0, BN, 32, HID);
  gemm_call(stream, EPI_TANH, true, m2h[0],KH, wM2[1], KH, pbM2[1], nullptr,0, m2h[1],KH, nullptr,0, BN, KH, HID);
  gemm_call(stream, EPI_TANH, true, m2h[1],KH, wM2[2], KH, pbM2[2], nullptr,0, m2h[2],KH, nullptr,0, BN, KH, HID);
  gemm_call(stream, EPI_LIN,  true, m2h[2],KH, wM2[3], KH, pbM2[3], nullptr,0, nullptr,0, o2f,64, BN, KH, 6);

  // ---- PSD assembly + 3x3 solves ----
  EL(k_psd_solve, BN, o1f, inp, 12, Ms1, pv);
  EL(k_psd_solve, BN, o2f, inp, 15, Ms2, pw);

  // ---- V net forward (hidden states only; head needed only for backward) ----
  gemm_call(stream, EPI_TANH, true, Apos, 64, wV[0], 64, pbV[0], nullptr,0, vh1,KH, nullptr,0, BATCH, 64, HID);
  gemm_call(stream, EPI_TANH, true, vh1, KH, wV[1], KH, pbV[1], nullptr,0, vh2,KH, nullptr,0, BATCH, KH, HID);

  // ---- g net forward + control force ----
  gemm_call(stream, EPI_TANH, true, Apos, 64, wG[0], 64, pbG[0], nullptr,0, gh1,KH, nullptr,0, BATCH, 64, HID);
  gemm_call(stream, EPI_TANH, true, gh1, KH, wG[1], KH, pbG[1], nullptr,0, gh2,KH, nullptr,0, BATCH, KH, HID);
  gemm_call(stream, EPI_LIN,  true, gh2, KH, wG[2], KH, pbG[2], nullptr,0, nullptr,0, gqf,192, BATCH, KH, 144);
  EL(k_force, BATCH*24, gqf, inp, Ff);

  // ---- V net backward: dV/dq ----
  EL(k_vnet_seed, BATCH*KH, vh2, vw[2], vseed);
  gemm_call(stream, EPI_TANHBWD, false, vseed,KH, tV[1], KH, nullptr, vh1,KH, vtmp,KH, nullptr,0, BATCH, KH, HID);
  gemm_call(stream, EPI_LIN,     false, vtmp, KH, tV[0], KH, nullptr, nullptr,0, nullptr,0, dVdq,64, BATCH, KH, 48);

  // ---- quadratic-form backward through M1, M2 (dH/dx, dH/dR PSD terms) ----
  EL(k_quad_seed, BN, o1f, pv, qseed);
  gemm_call(stream, EPI_TANHBWD, false, qseed,32, tM1[3], 32, nullptr, m1h[2],KH, bt1,KH, nullptr,0, BN, 32, HID);
  gemm_call(stream, EPI_TANHBWD, false, bt1,  KH, tM1[2], KH, nullptr, m1h[1],KH, bt2,KH, nullptr,0, BN, KH, HID);
  gemm_call(stream, EPI_TANHBWD, false, bt2,  KH, tM1[1], KH, nullptr, m1h[0],KH, bt1,KH, nullptr,0, BN, KH, HID);
  gemm_call(stream, EPI_LIN,     false, bt1,  KH, tM1[0], KH, nullptr, nullptr,0, nullptr,0, q1f,32, BN, KH, 3);
  EL(k_quad_seed, BN, o2f, pw, qseed);
  gemm_call(stream, EPI_TANHBWD, false, qseed,32, tM2[3], 32, nullptr, m2h[2],KH, bt1,KH, nullptr,0, BN, 32, HID);
  gemm_call(stream, EPI_TANHBWD, false, bt1,  KH, tM2[2], KH, nullptr, m2h[1],KH, bt2,KH, nullptr,0, BN, KH, HID);
  gemm_call(stream, EPI_TANHBWD, false, bt2,  KH, tM2[1], KH, nullptr, m2h[0],KH, bt1,KH, nullptr,0, BN, KH, HID);
  gemm_call(stream, EPI_LIN,     false, bt1,  KH, tM2[0], KH, nullptr, nullptr,0, nullptr,0, q2f,64, BN, KH, 9);

  // ---- dynamics assembly (dH/dpv = vs, dH/dpw = ws analytically) ----
  EL(k_combine, BN, dVdq, q1f, q2f, dHdx, dHdR);
  EL(k_dyn1, BN, inp, dxf, dRf, dxA, dRA);
  EL(k_dpvw, BN, inp, pv, pw, dHdx, dHdR, Ff, dpv, dpw);

  // ---- JVP of _psd(M1) along dx, then dv ----
  gemm_call(stream, EPI_TANHBWD, false, dxA,32, wM1[0], 32, nullptr, m1h[0],KH, bt1,KH, nullptr,0, BN, 32, HID);
  gemm_call(stream, EPI_TANHBWD, false, bt1,KH, wM1[1], KH, nullptr, m1h[1],KH, bt2,KH, nullptr,0, BN, KH, HID);
  gemm_call(stream, EPI_TANHBWD, false, bt2,KH, wM1[2], KH, nullptr, m1h[2],KH, bt1,KH, nullptr,0, BN, KH, HID);
  gemm_call(stream, EPI_LIN,     false, bt1,KH, wM1[3], KH, nullptr, nullptr,0, nullptr,0, tof,64, BN, KH, 6);
  EL(k_jvp_fin, BN, tof, o1f, Ms1, dpv, pv, dvf);

  // ---- JVP of _psd(M2) along dR, then dw ----
  gemm_call(stream, EPI_TANHBWD, false, dRA,32, wM2[0], 32, nullptr, m2h[0],KH, bt1,KH, nullptr,0, BN, 32, HID);
  gemm_call(stream, EPI_TANHBWD, false, bt1,KH, wM2[1], KH, nullptr, m2h[1],KH, bt2,KH, nullptr,0, BN, KH, HID);
  gemm_call(stream, EPI_TANHBWD, false, bt2,KH, wM2[2], KH, nullptr, m2h[2],KH, bt1,KH, nullptr,0, BN, KH, HID);
  gemm_call(stream, EPI_LIN,     false, bt1,KH, wM2[3], KH, nullptr, nullptr,0, nullptr,0, tof,64, BN, KH, 6);
  EL(k_jvp_fin, BN, tof, o2f, Ms2, dpw, pw, dwf);

  // ---- final output ----
  EL(k_assemble, BATCH*ROW, dxf, dRf, dvf, dwf, out);
}
